// KroneckerMoE_90580860273175
// MI455X (gfx1250) — compile-verified, matrix operand order checked by reading, not run on hardware
//
#include <hip/hip_runtime.h>
#include <hip/hip_bf16.h>

#define DIN   4096
#define NEXP  128
#define DSUB  64
#define TOPK  2
#define LSTR  80   // bf16 LDS row stride (160 B -> 16B-aligned vector rows)

typedef __attribute__((ext_vector_type(2)))  float  v2f;
typedef __attribute__((ext_vector_type(8)))  float  v8f;
typedef __attribute__((ext_vector_type(8)))  __bf16 v8bf;
typedef __attribute__((ext_vector_type(16))) __bf16 v16bf;
typedef int v4i __attribute__((vector_size(16)));          // matches builtin param
typedef __attribute__((address_space(1))) v4i* gv4i_p;     // global int4*
typedef __attribute__((address_space(3))) v4i* lv4i_p;     // LDS int4*

// split a float into bf16 hi + bf16 residual (error-compensated bf16x2)
__device__ __forceinline__ void bfsplit(float f, __bf16& h, __bf16& l) {
  h = (__bf16)f;
  l = (__bf16)(f - (float)h);
}

// Load a 16-bit WMMA fragment half-pair: K = {0..7} at p, K = {16..23} at p+16.
// Two 16-byte loads (global_load_b128 / ds_load_b128) + register concat.
__device__ __forceinline__ v16bf ldfrag(const __bf16* p) {
  v8bf a = *(const v8bf*)p;
  v8bf b = *(const v8bf*)(p + 16);
  return __builtin_shufflevector(a, b, 0, 1, 2, 3, 4, 5, 6, 7,
                                 8, 9, 10, 11, 12, 13, 14, 15);
}

// ---------------------------------------------------------------------------
// Kernel 0: pre-split expert weights into bf16 hi/lo planes (row-major, same
// indexing as the f32 originals). Runs once per launch; amortized over the
// ~128 tokens that hit each expert.
// ---------------------------------------------------------------------------
__global__ __launch_bounds__(256) void krmoe_split_w(
    const float* __restrict__ Aex, const float* __restrict__ Bex,
    __bf16* __restrict__ Ah, __bf16* __restrict__ Al,
    __bf16* __restrict__ Bh, __bf16* __restrict__ Bl, int total)
{
  int i = blockIdx.x * 256 + threadIdx.x;
  if (i < total) {
    __bf16 h, l;
    bfsplit(Aex[i], h, l); Ah[i] = h; Al[i] = l;
    bfsplit(Bex[i], h, l); Bh[i] = h; Bl[i] = l;
  }
}

// ---------------------------------------------------------------------------
// Kernel 1: router. logits[16 tok x 128 exp] per block via f32 WMMA 16x16x4,
// exact fp32 so top-k selection matches the fp32 reference. Then top-2 +
// softmax -> meta arrays in workspace.
// ---------------------------------------------------------------------------
__global__ __launch_bounds__(256) void krmoe_router(
    const float* __restrict__ x, const float* __restrict__ rw, int N,
    int* __restrict__ meta_idx, float* __restrict__ meta_p)
{
  __shared__ float lgt[16][132];          // padded: scanning threads hit distinct banks
  const int tid  = threadIdx.x;
  const int wave = tid >> 5;
  const int lane = tid & 31;
  const int half = lane >> 4;             // K sub-slot: lanes 16-31 hold K+2
  const int l15  = lane & 15;
  const int tok0 = blockIdx.x * 16;

  int tok = tok0 + l15; if (tok >= N) tok = N - 1;
  const float* xr = x  + (size_t)tok * DIN + 2 * half;
  const float* br = rw + (size_t)(wave * 16 + l15) * DIN + 2 * half;

  v8f acc = {};
  for (int k0 = 0; k0 < DIN; k0 += 16) {
#pragma unroll
    for (int kk = 0; kk < 16; kk += 4) {
      v2f a = *(const v2f*)(xr + k0 + kk);
      v2f b = *(const v2f*)(br + k0 + kk);
      acc = __builtin_amdgcn_wmma_f32_16x16x4_f32(false, a, false, b,
                                                  (short)0, acc, false, false);
    }
  }
#pragma unroll
  for (int r = 0; r < 8; ++r)
    lgt[r + 8 * half][wave * 16 + l15] = acc[r];
  __syncthreads();

  if (tid < 16) {
    int t = tok0 + tid;
    if (t < N) {
      float v0 = -3.4e38f, v1 = -3.4e38f; int i0 = 0, i1 = 0;
      for (int e = 0; e < NEXP; ++e) {
        float v = lgt[tid][e];
        if (v > v0)      { v1 = v0; i1 = i0; v0 = v; i0 = e; }
        else if (v > v1) { v1 = v;  i1 = e; }
      }
      float p0 = 1.f / (1.f + expf(v1 - v0));   // stable: v1 <= v0
      meta_idx[2 * t + 0] = i0;  meta_idx[2 * t + 1] = i1;
      meta_p  [2 * t + 0] = p0;  meta_p  [2 * t + 1] = 1.f - p0;
    }
  }
}

// ---------------------------------------------------------------------------
// Kernel 2: one block (8 waves) per token.
//   T = A_e @ X          (X^T staged split-bf16 in LDS)
//   Y += (p*T) @ B_e^T   (p*T restaged split-bf16 in LDS; accumulates straight
//                         into the persistent Y tiles via the WMMA C operand)
// Each product: 3x v_wmma_f32_16x16x32_bf16 (hi*hi + hi*lo + lo*hi), f32 acc.
// x row is staged into LDS with the async global->LDS path (ASYNCcnt); the
// raw-f32 staging buffer aliases the T hi/lo region (disjoint phases).
// ---------------------------------------------------------------------------
__global__ __launch_bounds__(256) void krmoe_apply(
    const float* __restrict__ x,
    const __bf16* __restrict__ Ah, const __bf16* __restrict__ Al,
    const __bf16* __restrict__ Bh, const __bf16* __restrict__ Bl,
    const float* __restrict__ scale, const float* __restrict__ bias,
    const int* __restrict__ meta_idx, const float* __restrict__ meta_p,
    float* __restrict__ out, int N)
{
  __shared__ __attribute__((aligned(16))) __bf16 xt_hi[64 * LSTR];
  __shared__ __attribute__((aligned(16))) __bf16 xt_lo[64 * LSTR];
  __shared__ __attribute__((aligned(16))) __bf16 tbuf [2 * 64 * LSTR]; // 20480 B
  __bf16* t_hi = tbuf;
  __bf16* t_lo = tbuf + 64 * LSTR;
  float*  xraw = (float*)tbuf;            // 16384 B, staging phase only

  const int n    = blockIdx.x;
  const int tid  = threadIdx.x;
  const int wave = tid >> 5;
  const int lane = tid & 31;
  const int half = lane >> 4;
  const int l15  = lane & 15;
  const int kb   = half * 8;              // fragment K base (16-bit A/B layout)

  const float* xn = x + (size_t)n * DIN;

  // ---- stage raw x row into LDS (async DMA path when available) ----
#if __has_builtin(__builtin_amdgcn_global_load_async_to_lds_b128)
  {
    gv4i_p gp = (gv4i_p)(uintptr_t)xn;
    lv4i_p lp = (lv4i_p)(uint32_t)(uintptr_t)xraw;
#pragma unroll
    for (int it = 0; it < 4; ++it) {
      int idx = it * 256 + tid;           // 16-byte units
      __builtin_amdgcn_global_load_async_to_lds_b128(gp + idx, lp + idx, 0, 0);
    }
  }
#if __has_builtin(__builtin_amdgcn_s_wait_asynccnt)
  __builtin_amdgcn_s_wait_asynccnt(0);
#else
  asm volatile("s_wait_asynccnt 0x0" ::: "memory");
#endif
#else
  {
    uint32_t lp = (uint32_t)(uintptr_t)xraw;
#pragma unroll
    for (int it = 0; it < 4; ++it) {
      int off = (it * 256 + tid) * 16;
      uint64_t ga = (uint64_t)(uintptr_t)xn + (uint64_t)off;
      asm volatile("global_load_async_to_lds_b128 %0, %1, off"
                   :: "v"(lp + off), "v"(ga) : "memory");
    }
    asm volatile("s_wait_asynccnt 0x0" ::: "memory");
  }
#endif
  __syncthreads();

  // ---- transform: LDS f32 -> split-bf16 X^T: xt[i2][i1] = x[i1*64 + i2] ----
  for (int i = tid; i < DIN / 4; i += 256) {
    float4 v = *(const float4*)(xraw + 4 * i);
    float vv[4] = {v.x, v.y, v.z, v.w};
    int e = 4 * i;
#pragma unroll
    for (int c = 0; c < 4; ++c) {
      int i1 = (e + c) >> 6, i2 = (e + c) & 63;
      __bf16 h, l; bfsplit(vv[c], h, l);
      xt_hi[i2 * LSTR + i1] = h;
      xt_lo[i2 * LSTR + i1] = l;
    }
  }
  __syncthreads();   // xraw reads done; tbuf may now be reused for T

  const int t0 = wave * 2;
  const int mo = t0 >> 2;                 // output row-tile (o1/16)
  const int no = t0 & 3;                  // first output col-tile; second = no+1

  v8f Y0 = {}, Y1 = {};
  const float s = scale[0];

  for (int kx = 0; kx < TOPK; ++kx) {
    const int   e = meta_idx[2 * n + kx];
    const float p = meta_p  [2 * n + kx];
    const __bf16* Aeh = Ah + (size_t)e * (DSUB * DSUB);
    const __bf16* Ael = Al + (size_t)e * (DSUB * DSUB);
    const __bf16* Beh = Bh + (size_t)e * (DSUB * DSUB);
    const __bf16* Bel = Bl + (size_t)e * (DSUB * DSUB);
    __builtin_prefetch(Aeh + tid * 16, 0, 3);   // global_prefetch_b8 (L2-resident)
    __builtin_prefetch(Beh + tid * 16, 0, 3);

    // ---------------- GEMM1: T = A_e @ X ----------------
    v8f T0 = {}, T1 = {};
#pragma unroll
    for (int k0 = 0; k0 < DSUB; k0 += 32) {
      const int aoff = (mo * 16 + l15) * DSUB + k0 + kb;
      v16bf a_hi = ldfrag(Aeh + aoff);
      v16bf a_lo = ldfrag(Ael + aoff);
      const int xoff = (no * 16 + l15) * LSTR + k0 + kb;
      v16bf b0h = ldfrag(xt_hi + xoff);
      v16bf b0l = ldfrag(xt_lo + xoff);
      v16bf b1h = ldfrag(xt_hi + xoff + 16 * LSTR);
      v16bf b1l = ldfrag(xt_lo + xoff + 16 * LSTR);
      T0 = __builtin_amdgcn_wmma_f32_16x16x32_bf16(false, a_hi, false, b0h, (short)0, T0, false, false);
      T0 = __builtin_amdgcn_wmma_f32_16x16x32_bf16(false, a_hi, false, b0l, (short)0, T0, false, false);
      T0 = __builtin_amdgcn_wmma_f32_16x16x32_bf16(false, a_lo, false, b0h, (short)0, T0, false, false);
      T1 = __builtin_amdgcn_wmma_f32_16x16x32_bf16(false, a_hi, false, b1h, (short)0, T1, false, false);
      T1 = __builtin_amdgcn_wmma_f32_16x16x32_bf16(false, a_hi, false, b1l, (short)0, T1, false, false);
      T1 = __builtin_amdgcn_wmma_f32_16x16x32_bf16(false, a_lo, false, b1h, (short)0, T1, false, false);
    }

    __syncthreads();   // prior expert's GEMM2 reads of t_* are done
    // store p*T (split) to LDS: folds routing prob into the second GEMM
#pragma unroll
    for (int r = 0; r < 8; ++r) {
      int m = mo * 16 + r + 8 * half;
      __bf16 h, l;
      bfsplit(p * T0[r], h, l);
      t_hi[m * LSTR + no * 16 + l15] = h;  t_lo[m * LSTR + no * 16 + l15] = l;
      bfsplit(p * T1[r], h, l);
      t_hi[m * LSTR + (no + 1) * 16 + l15] = h;  t_lo[m * LSTR + (no + 1) * 16 + l15] = l;
    }
    __syncthreads();

    // ---------------- GEMM2: Y += (p*T) @ B_e^T ----------------
#pragma unroll
    for (int k0 = 0; k0 < DSUB; k0 += 32) {
      const int toff = (mo * 16 + l15) * LSTR + k0 + kb;
      v16bf a_hi = ldfrag(t_hi + toff);
      v16bf a_lo = ldfrag(t_lo + toff);
      const int boff = (no * 16 + l15) * DSUB + k0 + kb;
      v16bf b0h = ldfrag(Beh + boff);
      v16bf b0l = ldfrag(Bel + boff);
      v16bf b1h = ldfrag(Beh + boff + 16 * DSUB);
      v16bf b1l = ldfrag(Bel + boff + 16 * DSUB);
      Y0 = __builtin_amdgcn_wmma_f32_16x16x32_bf16(false, a_hi, false, b0h, (short)0, Y0, false, false);
      Y0 = __builtin_amdgcn_wmma_f32_16x16x32_bf16(false, a_hi, false, b0l, (short)0, Y0, false, false);
      Y0 = __builtin_amdgcn_wmma_f32_16x16x32_bf16(false, a_lo, false, b0h, (short)0, Y0, false, false);
      Y1 = __builtin_amdgcn_wmma_f32_16x16x32_bf16(false, a_hi, false, b1h, (short)0, Y1, false, false);
      Y1 = __builtin_amdgcn_wmma_f32_16x16x32_bf16(false, a_hi, false, b1l, (short)0, Y1, false, false);
      Y1 = __builtin_amdgcn_wmma_f32_16x16x32_bf16(false, a_lo, false, b1h, (short)0, Y1, false, false);
    }
  }

  // ---- epilogue: out = Y * scale + bias ----
#pragma unroll
  for (int r = 0; r < 8; ++r) {
    int o1 = mo * 16 + r + 8 * half;
    int c0 = o1 * DSUB + no * 16 + l15;
    int c1 = c0 + 16;
    out[(size_t)n * DIN + c0] = Y0[r] * s + bias[c0];
    out[(size_t)n * DIN + c1] = Y1[r] * s + bias[c1];
  }
  if (n == 0 && tid == 0) out[(size_t)N * DIN] = 0.f;   // aux_loss
}

// ---------------------------------------------------------------------------
extern "C" void kernel_launch(void* const* d_in, const int* in_sizes, int n_in,
                              void* d_out, int out_size, void* d_ws, size_t ws_size,
                              hipStream_t stream)
{
  const float* x     = (const float*)d_in[0];
  const float* rw    = (const float*)d_in[1];
  const float* Aex   = (const float*)d_in[2];
  const float* Bex   = (const float*)d_in[3];
  const float* scale = (const float*)d_in[4];
  const float* bias  = (const float*)d_in[5];
  float* out = (float*)d_out;

  const int N = in_sizes[0] / DIN;        // 8192 tokens
  const int W = NEXP * DSUB * DSUB;       // 524288 weights per expert tensor

  // workspace layout (all sections 16B-aligned)
  char* ws = (char*)d_ws;
  int*    meta_idx = (int*)ws;                       ws += (size_t)2 * N * sizeof(int);
  float*  meta_p   = (float*)ws;                     ws += (size_t)2 * N * sizeof(float);
  __bf16* Ah = (__bf16*)ws;  ws += (size_t)W * sizeof(__bf16);
  __bf16* Al = (__bf16*)ws;  ws += (size_t)W * sizeof(__bf16);
  __bf16* Bh = (__bf16*)ws;  ws += (size_t)W * sizeof(__bf16);
  __bf16* Bl = (__bf16*)ws;

  krmoe_split_w<<<(W + 255) / 256, 256, 0, stream>>>(Aex, Bex, Ah, Al, Bh, Bl, W);
  krmoe_router <<<(N + 15) / 16, 256, 0, stream>>>(x, rw, N, meta_idx, meta_p);
  krmoe_apply  <<<N, 256, 0, stream>>>(x, Ah, Al, Bh, Bl, scale, bias,
                                       meta_idx, meta_p, out, N);
}
